// TemporalAttention_75668733821247
// MI455X (gfx1250) — compile-verified, matrix operand order checked by reading, not run on hardware
//
#include <hip/hip_runtime.h>
#include <hip/hip_bf16.h>
#include <math.h>

// ---------------------------------------------------------------------------
// TemporalAttention for MI455X (gfx1250, wave32, WMMA 16x16x32 f16)
//
// Sizes: N=32768 nodes, K=16 neighbors, MEM=172, TIME=100, MSG=172, ATTN=128
//        KV_IN=444 (pad 448), QIN=272 (pad 288), OIN=300 (pad 320)
//
// All GEMMs via v_wmma_f32_16x16x32_f16. A-operands staged in LDS in the
// documented 16-bit A-matrix lane layout; B-operands (weights) pre-swizzled
// once into workspace so each lane loads a contiguous 32B fragment.
// ---------------------------------------------------------------------------

typedef __attribute__((ext_vector_type(16))) _Float16 v16h;
typedef __attribute__((ext_vector_type(8)))  float    v8f;

#define NN    32768
#define KNB   16
#define MEMD  172
#define TIMED 100
#define MSGD  172
#define ATTND 128
#define KVP   448   // 14 k-steps of 32
#define QINP  288   // 9  k-steps
#define OINP  320   // 10 k-steps
#define RSCALE 0.08838834764831845f  // 1/sqrt(128)

// workspace layout (bytes), all 256B aligned
#define OFF_Q    ((size_t)0)                         // N*128 f32   = 16,777,216
#define OFF_W    ((size_t)16777216)                  // N*448 f16   = 29,360,128
#define OFF_SAT  ((size_t)46137344)                  // N     f32   =    131,072
#define OFF_WKH  ((size_t)46268416)                  // 448*128 f16 =    114,688
#define OFF_WVH  ((size_t)46383104)                  // 448*128 f16
#define OFF_WQH  ((size_t)46497792)                  // 288*128 f16 =     73,728
#define OFF_WOH  ((size_t)46571520)                  // 320*128 f16 =     81,920

// WMMA 16-bit A/B operand swizzle (ISA 7.12.2): element (m,k) -> half index
// within a [ksteps*32 lanes * 16 halves] region.
__device__ __forceinline__ int swz(int m, int k) {
  int ks = k >> 5;
  int kk = k & 31;
  int hi = (kk >> 3) & 1;
  int v  = (kk < 16) ? ((kk & 7) >> 1) : (4 + ((kk & 7) >> 1));
  int lane = m + (hi << 4);
  int half = (v << 1) | (k & 1);
  return ((ks << 5) + lane) * 16 + half;
}

__device__ __forceinline__ v8f wmma16(v16h a, v16h b, v8f c) {
  return __builtin_amdgcn_wmma_f32_16x16x32_f16(false, a, false, b,
                                                (short)0, c, false, false);
}

__device__ __forceinline__ float red16_sum(float v) {
  v += __shfl_xor(v, 1, 32);
  v += __shfl_xor(v, 2, 32);
  v += __shfl_xor(v, 4, 32);
  v += __shfl_xor(v, 8, 32);
  return v;
}
__device__ __forceinline__ float red16_max(float v) {
  v = fmaxf(v, __shfl_xor(v, 1, 32));
  v = fmaxf(v, __shfl_xor(v, 2, 32));
  v = fmaxf(v, __shfl_xor(v, 4, 32));
  v = fmaxf(v, __shfl_xor(v, 8, 32));
  return v;
}

// ---------------------------------------------------------------------------
// K0: convert + pre-swizzle weights into WMMA B-operand order.
// Region layout per weight: [ntile(8)][kstep][32 lanes][16 halves].
// ---------------------------------------------------------------------------
__device__ __forceinline__ void packw(_Float16* dst, const float* src,
                                      int KP, int KS, int sid) {
  int k = sid >> 7;       // row
  int n = sid & 127;      // col
  float v = (k < KS) ? src[k * 128 + n] : 0.0f;
  int nt = n >> 4;
  dst[nt * ((KP >> 5) << 9) + swz(n & 15, k)] = (_Float16)v;
}

__global__ void ta_k0_pack(const float* wq, const float* wk, const float* wv,
                           const float* wo, _Float16* wkh, _Float16* wvh,
                           _Float16* wqh, _Float16* woh) {
  int id = blockIdx.x * 256 + threadIdx.x;
  const int E_K = KVP * 128;    // 57344
  const int E_Q = QINP * 128;   // 36864
  const int E_O = OINP * 128;   // 40960
  if (id < E_K)                      { packw(wkh, wk, KVP, 444, id); return; }
  id -= E_K;
  if (id < E_K)                      { packw(wvh, wv, KVP, 444, id); return; }
  id -= E_K;
  if (id < E_Q)                      { packw(wqh, wq, QINP, 272, id); return; }
  id -= E_Q;
  if (id < E_O)                      { packw(woh, wo, OINP, 300, id); return; }
}

// ---------------------------------------------------------------------------
// K1: q[N,128] = [memory_u | cos(te_b)] @ Wq + bq.   16-node tile per block,
// 8 waves = 8 N-tiles, 9 k-steps each.
// ---------------------------------------------------------------------------
__global__ void ta_k1_q(const float* __restrict__ memu,
                        const float* __restrict__ te_b,
                        const float* __restrict__ wq_b,
                        const _Float16* __restrict__ wqh,
                        float* __restrict__ qout) {
  __shared__ __align__(32) _Float16 aQ[9 * 512];
  const int tid = threadIdx.x;
  const int base = blockIdx.x * 16;

  for (int idx = tid; idx < 16 * QINP; idx += 256) {
    int m = idx / QINP, j = idx - m * QINP;
    float v;
    if (j < MEMD)      v = memu[(size_t)(base + m) * MEMD + j];
    else if (j < 272)  v = cosf(te_b[j - MEMD]);
    else               v = 0.0f;
    aQ[swz(m, j)] = (_Float16)v;
  }
  __syncthreads();

  const int wid = tid >> 5, lane = tid & 31, hi = lane >> 4, col = lane & 15;
  const int n0 = wid * 16;
  v8f acc = {};
#pragma unroll
  for (int ks = 0; ks < 9; ++ks) {
    v16h a = *(const v16h*)&aQ[(ks * 32 + lane) * 16];
    v16h b = *(const v16h*)&wqh[(wid * 9 + ks) * 512 + lane * 16];
    acc = wmma16(a, b, acc);
  }
  const float bias = wq_b[n0 + col];
#pragma unroll
  for (int r = 0; r < 8; ++r) {
    int row = r + 8 * hi;
    qout[(size_t)(base + row) * 128 + n0 + col] = acc[r] + bias;
  }
}

// ---------------------------------------------------------------------------
// K2: per node: build masked kv_in[16,448] in LDS, K-projection via WMMA
// (B fragments cached in VGPRs, reused across 16 nodes), scores -> softmax ->
// w = attn^T @ kv_in  (f16 to workspace), plus sum(attn) flag.
// ---------------------------------------------------------------------------
__global__ void ta_k2_attn(const float* __restrict__ tref,
                           const float* __restrict__ nbrt,
                           const float* __restrict__ nbrmem,
                           const float* __restrict__ nbrmsg,
                           const unsigned char* __restrict__ valid, // bool[N,K]
                           const float* __restrict__ te_w,
                           const float* __restrict__ te_b,
                           const _Float16* __restrict__ wkh,
                           const float* __restrict__ qin,
                           _Float16* __restrict__ wout,
                           float* __restrict__ sattn_out) {
  __shared__ __align__(32) _Float16 aKV[14 * 512];
  __shared__ float dtv[16], mk[16], sc[16], at[16];

  const int tid = threadIdx.x;
  const int wid = tid >> 5, lane = tid & 31, hi = lane >> 4, col = lane & 15;
  const int n0 = wid * 16;

  // Cache this wave's 14 B fragments (K-projection weights) in registers.
  v16h bfr[14];
#pragma unroll
  for (int ks = 0; ks < 14; ++ks)
    bfr[ks] = *(const v16h*)&wkh[(wid * 14 + ks) * 512 + lane * 16];

  const int base = blockIdx.x * 16;
  for (int it = 0; it < 16; ++it) {
    const int node = base + it;

    if (tid < 16) {
      float dt = tref[node] - nbrt[(size_t)node * KNB + tid];
      dtv[tid] = fmaxf(dt, 0.0f);
      mk[tid]  = valid[(size_t)node * KNB + tid] ? 1.0f : 0.0f;
      sc[tid]  = 0.0f;
    }
    __syncthreads();

    // Build kv_in (masked) straight into the swizzled A layout.
    for (int idx = tid; idx < 16 * KVP; idx += 256) {
      int kn = idx / KVP, j = idx - kn * KVP;
      float v;
      if (j < MEMD)      v = nbrmem[((size_t)node * KNB + kn) * MEMD + j];
      else if (j < 272)  { int t = j - MEMD; v = cosf(dtv[kn] * te_w[t] + te_b[t]); }
      else if (j < 444)  v = nbrmsg[((size_t)node * KNB + kn) * MSGD + (j - 272)];
      else               v = 0.0f;
      aKV[swz(kn, j)] = (_Float16)(v * mk[kn]);
    }
    __syncthreads();

    // K projection: k-tile [16 neighbors x 16 cols] per wave.
    v8f acc = {};
#pragma unroll
    for (int ks = 0; ks < 14; ++ks) {
      v16h a = *(const v16h*)&aKV[(ks * 32 + lane) * 16];
      acc = wmma16(a, bfr[ks], acc);
    }

    // scores[kn] += sum over this wave's 16 cols of k * q   (bias term q.bk
    // is constant over kn -> softmax-invariant, skipped).
    const float qv = qin[(size_t)node * 128 + n0 + col];
#pragma unroll
    for (int r = 0; r < 8; ++r) {
      float p = red16_sum(acc[r] * qv);
      if (col == 0) atomicAdd(&sc[r + 8 * hi], p);
    }
    __syncthreads();

    // Masked softmax over the 16 neighbors (wave 0, lanes 0..15).
    if (tid < 16) {
      float anyv = red16_max(mk[tid]);
      float a_;
      float s = (mk[tid] > 0.0f) ? sc[tid] * RSCALE : -INFINITY;
      if (anyv > 0.0f) {
        float mx = red16_max(s);
        float e  = (mk[tid] > 0.0f) ? expf(s - mx) : 0.0f;
        float sm = red16_sum(e);
        a_ = e / sm;
      } else {
        a_ = 0.0f;
      }
      at[tid] = a_;
      if (tid == 0) sattn_out[node] = (anyv > 0.0f) ? 1.0f : 0.0f;
    }
    __syncthreads();

    // w[j] = sum_kn attn[kn] * kv_in[kn][j]   (row-major f16 to workspace)
    for (int j = tid; j < KVP; j += 256) {
      float accw = 0.0f;
#pragma unroll
      for (int kn = 0; kn < 16; ++kn)
        accw += at[kn] * (float)aKV[swz(kn, j)];
      wout[(size_t)node * KVP + j] = (_Float16)accw;
    }
    __syncthreads();
  }
}

// ---------------------------------------------------------------------------
// K3: z = w @ Wv + sattn*bv ; out = [memory_u | z] @ Wo + bo ; LayerNorm.
// 16-node tile per block, fully fused.
// ---------------------------------------------------------------------------
__global__ void ta_k3_out(const float* __restrict__ memu,
                          const _Float16* __restrict__ win,
                          const float* __restrict__ sattn_in,
                          const _Float16* __restrict__ wvh,
                          const float* __restrict__ wv_b,
                          const _Float16* __restrict__ woh,
                          const float* __restrict__ wo_b,
                          const float* __restrict__ ln_g,
                          const float* __restrict__ ln_b,
                          float* __restrict__ out) {
  __shared__ __align__(32) _Float16 aW[14 * 512];
  __shared__ __align__(32) _Float16 aO[10 * 512];
  __shared__ float sat[16], sm[16], sq[16];

  const int tid = threadIdx.x;
  const int wid = tid >> 5, lane = tid & 31, hi = lane >> 4, col = lane & 15;
  const int n0 = wid * 16;
  const int base = blockIdx.x * 16;

  if (tid < 16) { sat[tid] = sattn_in[base + tid]; sm[tid] = 0.0f; sq[tid] = 0.0f; }

  for (int idx = tid; idx < 16 * KVP; idx += 256) {
    int m = idx / KVP, j = idx - m * KVP;
    aW[swz(m, j)] = win[(size_t)(base + m) * KVP + j];
  }
  // memory_u part + pad of the output-GEMM A operand (z cols filled by waves).
  for (int idx = tid; idx < 16 * OINP; idx += 256) {
    int m = idx / OINP, j = idx - m * OINP;
    if (j < MEMD)      aO[swz(m, j)] = (_Float16)memu[(size_t)(base + m) * MEMD + j];
    else if (j >= 300) aO[swz(m, j)] = (_Float16)0.0f;
  }
  __syncthreads();

  // z projection
  v8f zacc = {};
#pragma unroll
  for (int ks = 0; ks < 14; ++ks) {
    v16h a = *(const v16h*)&aW[(ks * 32 + lane) * 16];
    v16h b = *(const v16h*)&wvh[(wid * 14 + ks) * 512 + lane * 16];
    zacc = wmma16(a, b, zacc);
  }
  const float vb = wv_b[n0 + col];
#pragma unroll
  for (int r = 0; r < 8; ++r) {
    int row = r + 8 * hi;
    float z = zacc[r] + sat[row] * vb;
    aO[swz(row, MEMD + n0 + col)] = (_Float16)z;
  }
  __syncthreads();

  // output projection
  v8f oacc = {};
#pragma unroll
  for (int ks = 0; ks < 10; ++ks) {
    v16h a = *(const v16h*)&aO[(ks * 32 + lane) * 16];
    v16h b = *(const v16h*)&woh[(wid * 10 + ks) * 512 + lane * 16];
    oacc = wmma16(a, b, oacc);
  }
  const float ob = wo_b[n0 + col];
  float x[8];
#pragma unroll
  for (int r = 0; r < 8; ++r) {
    x[r] = oacc[r] + ob;
    float p  = red16_sum(x[r]);
    float pp = red16_sum(x[r] * x[r]);
    if (col == 0) { atomicAdd(&sm[r + 8 * hi], p); atomicAdd(&sq[r + 8 * hi], pp); }
  }
  __syncthreads();

  const float g = ln_g[n0 + col], bb = ln_b[n0 + col];
#pragma unroll
  for (int r = 0; r < 8; ++r) {
    int row = r + 8 * hi;
    float mu  = sm[row] * (1.0f / 128.0f);
    float var = sq[row] * (1.0f / 128.0f) - mu * mu;
    float rs  = rsqrtf(var + 1e-5f);
    out[(size_t)(base + row) * 128 + n0 + col] = (x[r] - mu) * rs * g + bb;
  }
}

// ---------------------------------------------------------------------------
extern "C" void kernel_launch(void* const* d_in, const int* in_sizes, int n_in,
                              void* d_out, int out_size, void* d_ws, size_t ws_size,
                              hipStream_t stream) {
  const float* memory_u = (const float*)d_in[0];
  const float* t_ref    = (const float*)d_in[1];
  const float* nbr_mem  = (const float*)d_in[2];
  const float* nbr_t    = (const float*)d_in[3];
  const float* nbr_msg  = (const float*)d_in[4];
  const unsigned char* valid = (const unsigned char*)d_in[5]; // bool array
  const float* te_w     = (const float*)d_in[6];
  const float* te_b     = (const float*)d_in[7];
  const float* wq_w     = (const float*)d_in[8];
  const float* wq_b     = (const float*)d_in[9];
  const float* wk_w     = (const float*)d_in[10];
  // wk_b (d_in[11]) is softmax-invariant -> unused.
  const float* wv_w     = (const float*)d_in[12];
  const float* wv_b     = (const float*)d_in[13];
  const float* wo_w     = (const float*)d_in[14];
  const float* wo_b     = (const float*)d_in[15];
  const float* ln_g     = (const float*)d_in[16];
  const float* ln_b     = (const float*)d_in[17];

  char* ws = (char*)d_ws;
  float*     q_ws   = (float*)    (ws + OFF_Q);
  _Float16*  w_ws   = (_Float16*) (ws + OFF_W);
  float*     sat_ws = (float*)    (ws + OFF_SAT);
  _Float16*  wkh    = (_Float16*) (ws + OFF_WKH);
  _Float16*  wvh    = (_Float16*) (ws + OFF_WVH);
  _Float16*  wqh    = (_Float16*) (ws + OFF_WQH);
  _Float16*  woh    = (_Float16*) (ws + OFF_WOH);

  // K0: pack/swizzle weights (192512 elements)
  ta_k0_pack<<<752, 256, 0, stream>>>(wq_w, wk_w, wv_w, wo_w, wkh, wvh, wqh, woh);
  // K1: q GEMM (2048 tiles of 16 nodes)
  ta_k1_q<<<NN / 16, 256, 0, stream>>>(memory_u, te_b, wq_b, wqh, q_ws);
  // K2: attention (2048 blocks x 16 nodes each)
  ta_k2_attn<<<NN / 256 * 16, 256, 0, stream>>>(t_ref, nbr_t, nbr_mem, nbr_msg,
                                                valid, te_w, te_b, wkh, q_ws,
                                                w_ws, sat_ws);
  // K3: fused z/out GEMM + LayerNorm
  ta_k3_out<<<NN / 16, 256, 0, stream>>>(memory_u, w_ws, sat_ws, wvh, wv_b,
                                         woh, wo_b, ln_g, ln_b, (float*)d_out);
}
